// WordEncoder_80058190397895
// MI455X (gfx1250) — matrix operand order repaired and checked by reading, not compiled, not run on hardware
//
#include <hip/hip_runtime.h>

// ---------------------------------------------------------------------------
// 2-layer bidirectional LSTM for MI455X (gfx1250, wave32, WMMA f16->f32).
// B=128, T=256, D=768, H=768.  gates = [x_t, h_{t-1}] @ Wcat^T + b
// Each wave computes one 16x16 (batch x hidden) block for ALL FOUR gates
// (4 f32 accumulators), then applies the LSTM cell in registers.
// K-loop is split into compile-time-sized x / h segments (branch-free,
// partially unrolled so K offsets become instruction immediates).
// ---------------------------------------------------------------------------

#define TT 256
#define BB 128
#define DD 768
#define HH 768
#define NG 3072   // 4*H

typedef _Float16 v16h __attribute__((ext_vector_type(16)));
typedef _Float16 v8h  __attribute__((ext_vector_type(8)));
typedef float    v8f  __attribute__((ext_vector_type(8)));

__device__ inline v16h cat8(v8h a, v8h b) {
  return __builtin_shufflevector(a, b, 0,1,2,3,4,5,6,7,8,9,10,11,12,13,14,15);
}
__device__ inline float sigf(float x)  { return 1.0f / (1.0f + __expf(-x)); }
__device__ inline float tanhf_(float x){ return 2.0f * sigf(2.0f * x) - 1.0f; }

#define WMMA16(A, B, C) \
  __builtin_amdgcn_wmma_f32_16x16x32_f16(false, (A), false, (B), (short)0, (C), false, false)

// ---- convert [W_ih ; W_hh] (f32) -> concatenated f16 (NG x K), row-major ----
__global__ void convert_w_kernel(const float* __restrict__ wih,
                                 const float* __restrict__ whh,
                                 _Float16* __restrict__ out,
                                 int din, int K) {
  size_t idx = (size_t)blockIdx.x * blockDim.x + threadIdx.x;
  size_t total = (size_t)NG * (size_t)K;
  if (idx >= total) return;
  int k = (int)(idx % (size_t)K);
  int n = (int)(idx / (size_t)K);
  float v = (k < din) ? wih[(size_t)n * din + k]
                      : whh[(size_t)n * HH + (k - din)];
  out[idx] = (_Float16)v;
}

// ---- zero h (slot 0, f16) and c (f32) before each directional scan ----
__global__ void init_state_kernel(_Float16* __restrict__ h0,
                                  float* __restrict__ c) {
  int i = blockIdx.x * blockDim.x + threadIdx.x;
  if (i < BB * HH) { h0[i] = (_Float16)0.0f; c[i] = 0.0f; }
}

// ---- one LSTM timestep: WMMA GEMM over K = DIN + H, fused cell ----
// XF32: layer-0 x source is (B,T,DD) f32; else (T,B,2H) f16 time-major.
// OUT16: write h to f16 (T,B,2H) buffer (layer-0); else f32 (B,T,2H) d_out.
template <bool XF32, bool OUT16, int DIN>
__global__ void __launch_bounds__(256)
lstm_step_kernel(const void* __restrict__ xsrc,
                 const _Float16* __restrict__ w,     // (NG, K) f16
                 const float* __restrict__ bias,     // (NG)
                 const float* __restrict__ mask,     // (B, T)
                 const _Float16* __restrict__ hprev, // (B, H) f16
                 _Float16* __restrict__ hnext,       // (B, H) f16
                 float* __restrict__ cbuf,           // (B, H) f32, in-place
                 _Float16* __restrict__ y16,         // (T, B, 2H) f16
                 float* __restrict__ yf32,           // (B, T, 2H) f32
                 int t_eff, int outbase) {
  constexpr int K = DIN + HH;
  const int wave  = threadIdx.x >> 5;
  const int lane  = threadIdx.x & 31;
  const bool lo   = lane < 16;
  const int col   = lane & 15;
  const int ntile = blockIdx.x * 8 + wave;     // 0..47 (H/16 = 48)
  const int n0    = ntile * 16;
  const int m0    = blockIdx.y * 16;           // 8 mtiles (B/16)
  const int mA    = m0 + col;                  // A-fragment row for this lane
  const int b0    = lo ? 0 : 16;               // A-frag K offsets (contiguous k = 16h + s)
  const int b1    = lo ? 8 : 24;
  const int bb    = lo ? 0 : 16;               // B-frag K offset

  // accumulators = bias (broadcast down M) for gates i,f,g,o
  v8f acc0, acc1, acc2, acc3;
  {
    float bv0 = bias[0 * HH + n0 + col];
    float bv1 = bias[1 * HH + n0 + col];
    float bv2 = bias[2 * HH + n0 + col];
    float bv3 = bias[3 * HH + n0 + col];
#pragma unroll
    for (int r = 0; r < 8; ++r) {
      acc0[r] = bv0; acc1[r] = bv1; acc2[r] = bv2; acc3[r] = bv3;
    }
  }

  const _Float16* wr0 = w + (size_t)(0 * HH + n0 + col) * K + bb;
  const _Float16* wr1 = w + (size_t)(1 * HH + n0 + col) * K + bb;
  const _Float16* wr2 = w + (size_t)(2 * HH + n0 + col) * K + bb;
  const _Float16* wr3 = w + (size_t)(3 * HH + n0 + col) * K + bb;

  // ---- segment 1: x contribution (K = 0 .. DIN) ----
  if (XF32) {
    const float* xr = (const float*)xsrc + (size_t)mA * TT * DD + (size_t)t_eff * DD;
#pragma unroll 4
    for (int kk = 0; kk < DIN; kk += 32) {
      v16h a;
#pragma unroll
      for (int j = 0; j < 8; ++j) {
        a[j]     = (_Float16)xr[kk + b0 + j];
        a[j + 8] = (_Float16)xr[kk + b1 + j];
      }
      v16h w0 = *(const v16h*)(wr0 + kk);
      v16h w1 = *(const v16h*)(wr1 + kk);
      v16h w2 = *(const v16h*)(wr2 + kk);
      v16h w3 = *(const v16h*)(wr3 + kk);
      acc0 = WMMA16(a, w0, acc0);
      acc1 = WMMA16(a, w1, acc1);
      acc2 = WMMA16(a, w2, acc2);
      acc3 = WMMA16(a, w3, acc3);
    }
  } else {
    const _Float16* xr = (const _Float16*)xsrc + (size_t)t_eff * BB * (2 * HH)
                                               + (size_t)mA * (2 * HH);
#pragma unroll 4
    for (int kk = 0; kk < DIN; kk += 32) {
      v16h a  = cat8(*(const v8h*)(xr + kk + b0), *(const v8h*)(xr + kk + b1));
      v16h w0 = *(const v16h*)(wr0 + kk);
      v16h w1 = *(const v16h*)(wr1 + kk);
      v16h w2 = *(const v16h*)(wr2 + kk);
      v16h w3 = *(const v16h*)(wr3 + kk);
      acc0 = WMMA16(a, w0, acc0);
      acc1 = WMMA16(a, w1, acc1);
      acc2 = WMMA16(a, w2, acc2);
      acc3 = WMMA16(a, w3, acc3);
    }
  }

  // ---- segment 2: recurrent contribution (K = DIN .. DIN+H) ----
  {
    const _Float16* hr = hprev + (size_t)mA * HH;
#pragma unroll 4
    for (int kk = 0; kk < HH; kk += 32) {
      v16h a  = cat8(*(const v8h*)(hr + kk + b0), *(const v8h*)(hr + kk + b1));
      v16h w0 = *(const v16h*)(wr0 + DIN + kk);
      v16h w1 = *(const v16h*)(wr1 + DIN + kk);
      v16h w2 = *(const v16h*)(wr2 + DIN + kk);
      v16h w3 = *(const v16h*)(wr3 + DIN + kk);
      acc0 = WMMA16(a, w0, acc0);
      acc1 = WMMA16(a, w1, acc1);
      acc2 = WMMA16(a, w2, acc2);
      acc3 = WMMA16(a, w3, acc3);
    }
  }

  // ---- fused LSTM cell: all four gates for (m, n) live in this wave ----
#pragma unroll
  for (int r = 0; r < 8; ++r) {
    int m = m0 + r + (lo ? 0 : 8);
    float mt = mask[(size_t)m * TT + t_eff];
    size_t ci = (size_t)m * HH + n0 + col;
    float ig = sigf(acc0[r]);
    float fg = sigf(acc1[r]);
    float gg = tanhf_(acc2[r]);
    float og = sigf(acc3[r]);
    float cn = fg * cbuf[ci] + ig * gg;
    float hn = og * tanhf_(cn);
    hn *= mt;
    cn *= mt;
    cbuf[ci] = cn;
    hnext[ci] = (_Float16)hn;
    if (OUT16) {
      y16[(size_t)t_eff * BB * (2 * HH) + (size_t)m * (2 * HH) + outbase + n0 + col] =
          (_Float16)hn;
    } else {
      yf32[(size_t)m * TT * (2 * HH) + (size_t)t_eff * (2 * HH) + outbase + n0 + col] = hn;
    }
  }
}

// ---------------------------------------------------------------------------
extern "C" void kernel_launch(void* const* d_in, const int* in_sizes, int n_in,
                              void* d_out, int out_size, void* d_ws, size_t ws_size,
                              hipStream_t stream) {
  (void)in_sizes; (void)n_in; (void)out_size; (void)ws_size;
  const float* x    = (const float*)d_in[0];   // (B,T,D)
  const float* mask = (const float*)d_in[1];   // (B,T)
  const float* wih[4]  = {(const float*)d_in[2], (const float*)d_in[5],
                          (const float*)d_in[8], (const float*)d_in[11]};
  const float* whh[4]  = {(const float*)d_in[3], (const float*)d_in[6],
                          (const float*)d_in[9], (const float*)d_in[12]};
  const float* bias[4] = {(const float*)d_in[4], (const float*)d_in[7],
                          (const float*)d_in[10], (const float*)d_in[13]};

  const int Ks[4]   = {DD + HH, DD + HH, 2 * HH + HH, 2 * HH + HH};  // 1536,1536,2304,2304
  const int dins[4] = {DD, DD, 2 * HH, 2 * HH};

  // workspace carving (256B aligned)
  char* ws = (char*)d_ws;
  size_t off = 0;
  auto carve = [&](size_t bytes) -> void* {
    void* p = ws + off;
    off += (bytes + 255) & ~(size_t)255;
    return p;
  };
  _Float16* y16 = (_Float16*)carve((size_t)TT * BB * (2 * HH) * sizeof(_Float16)); // 96 MB
  _Float16* w16[4];
  for (int s = 0; s < 4; ++s)
    w16[s] = (_Float16*)carve((size_t)NG * Ks[s] * sizeof(_Float16));              // 47 MB
  _Float16* h16 = (_Float16*)carve((size_t)2 * BB * HH * sizeof(_Float16));        // ping-pong
  float* cbuf   = (float*)carve((size_t)BB * HH * sizeof(float));

  // one-time weight conversion to f16 (stays resident in 192MB L2 thereafter)
  for (int s = 0; s < 4; ++s) {
    size_t tot = (size_t)NG * Ks[s];
    convert_w_kernel<<<dim3((unsigned)((tot + 255) / 256)), dim3(256), 0, stream>>>(
        wih[s], whh[s], w16[s], dins[s], Ks[s]);
  }

  const dim3 grid(6, 8);   // 48 ntiles x 8 mtiles = 384 waves/step
  const dim3 blk(256);
  const int initN = (BB * HH + 255) / 256;

  for (int s = 0; s < 4; ++s) {            // L0 fwd, L0 bwd, L1 fwd, L1 bwd
    const bool bwd = (s & 1);
    const bool l1  = (s >= 2);
    init_state_kernel<<<dim3(initN), dim3(256), 0, stream>>>(h16, cbuf);
    for (int st = 0; st < TT; ++st) {
      const int t_eff = bwd ? (TT - 1 - st) : st;
      _Float16* hp = h16 + (size_t)(st & 1) * (BB * HH);
      _Float16* hn = h16 + (size_t)((st + 1) & 1) * (BB * HH);
      if (!l1) {
        lstm_step_kernel<true, true, DD><<<grid, blk, 0, stream>>>(
            (const void*)x, w16[s], bias[s], mask, hp, hn, cbuf,
            y16, nullptr, t_eff, bwd ? HH : 0);
      } else {
        lstm_step_kernel<false, false, 2 * HH><<<grid, blk, 0, stream>>>(
            (const void*)y16, w16[s], bias[s], mask, hp, hn, cbuf,
            nullptr, (float*)d_out, t_eff, bwd ? HH : 0);
      }
    }
  }
}
